// VarianceAdaptor_11304353923427
// MI455X (gfx1250) — compile-verified, matrix operand order checked by reading, not run on hardware
//
#include <hip/hip_runtime.h>
#include <hip/hip_bf16.h>
#include <math.h>

// ---------------- problem constants ----------------
#define B_DIM 32
#define S_DIM 256
#define M_DIM 2048
#define H_DIM 384      // hidden == filter == 384
#define C_DIM 384
#define NB_DIM 256
#define LN_EPS 1e-5f

typedef __attribute__((ext_vector_type(16))) _Float16 v16h;
typedef __attribute__((ext_vector_type(8)))  float    v8f;

__device__ __forceinline__ float wave_red(float v) {
    #pragma unroll
    for (int o = 16; o > 0; o >>= 1) v += __shfl_xor(v, o, 32);
    return v;
}

__device__ __forceinline__ void load_frag(v16h& d, const _Float16* p) {
    uint4* q = reinterpret_cast<uint4*>(&d);
    q[0] = *reinterpret_cast<const uint4*>(p);
    q[1] = *reinterpret_cast<const uint4*>(p + 16);
}

// ---------------- prep: x f32 -> padded f16 [B][T+2][C] ----------------
__global__ void cvt_pad_kernel(const float* __restrict__ x, _Float16* __restrict__ xb, int T) {
    const int n = B_DIM * (T + 2) * C_DIM;
    int i = blockIdx.x * blockDim.x + threadIdx.x;
    if (i >= n) return;
    const int c = i % C_DIM;
    const int t = (i / C_DIM) % (T + 2);
    const int b = i / (C_DIM * (T + 2));
    _Float16 v = (_Float16)0.f;
    if (t != 0 && t != T + 1)
        v = (_Float16)x[((size_t)b * T + (t - 1)) * C_DIM + c];
    xb[i] = v;
}

// w[3][Cin][F] f32  ->  wT[3][F][Cin] f16
__global__ void transpose_w_kernel(const float* __restrict__ w, _Float16* __restrict__ wT) {
    int i = blockIdx.x * blockDim.x + threadIdx.x;
    const int n = 3 * C_DIM * C_DIM;
    if (i >= n) return;
    int tap = i / (C_DIM * C_DIM);
    int rem = i % (C_DIM * C_DIM);
    int f = rem / C_DIM;
    int c = rem % C_DIM;
    wT[i] = (_Float16)w[(size_t)tap * C_DIM * C_DIM + (size_t)c * C_DIM + f];
}

// ---------------- fused conv(k=3) + bias + LN + ReLU ----------------
// in  : padded f16 [B][T+2][C]
// out : padded f16 [B][T+2][C]   (this kernel also writes its own pad rows)
// block: 768 threads = 24 waves; computes 32 time rows x 384 features.
// each wave: two 16x16 tiles (t-rows 0-15 / 16-31) sharing each B fragment.
__global__ __launch_bounds__(768)
void conv_ln_relu_kernel(const _Float16* __restrict__ in, const _Float16* __restrict__ wT,
                         const float* __restrict__ cbias, const float* __restrict__ lng,
                         const float* __restrict__ lnb, _Float16* __restrict__ out, int T)
{
    __shared__ float tile[32 * C_DIM];
    __shared__ float sMu[32], sRs[32];
    const int tid   = threadIdx.x;
    const int wave  = tid >> 5;          // 0..23
    const int lane  = tid & 31;
    const int laneM = lane & 15;
    const int half  = lane >> 4;
    const int nb    = T / 32;
    const int b     = blockIdx.x / nb;
    const int t0    = (blockIdx.x % nb) * 32;
    const int f0    = wave * 16;
    const int Tp    = T + 2;

    v8f acc0 = {}, acc1 = {};
    #pragma unroll
    for (int tap = 0; tap < 3; ++tap) {
        const _Float16* pa0 = in + ((size_t)b * Tp + t0 + laneM + tap) * C_DIM + half * 8;
        const _Float16* pa1 = pa0 + (size_t)16 * C_DIM;
        const _Float16* pb  = wT + ((size_t)tap * C_DIM + f0 + laneM) * C_DIM + half * 8;
        for (int cb = 0; cb < C_DIM; cb += 32) {
            v16h a0, a1, bf;
            load_frag(a0, pa0 + cb);
            load_frag(a1, pa1 + cb);
            load_frag(bf, pb + cb);
            acc0 = __builtin_amdgcn_wmma_f32_16x16x32_f16(false, a0, false, bf,
                                                          (short)0, acc0, false, false);
            acc1 = __builtin_amdgcn_wmma_f32_16x16x32_f16(false, a1, false, bf,
                                                          (short)0, acc1, false, false);
        }
    }
    // stage tiles + conv bias into LDS (VGPR r -> M = r + 8*half, N = laneM)
    {
        const int fcol = f0 + laneM;
        const float cb_ = cbias[fcol];
        #pragma unroll
        for (int r = 0; r < 8; ++r) {
            const int m = r + half * 8;
            tile[m * C_DIM + fcol]        = acc0[r] + cb_;
            tile[(m + 16) * C_DIM + fcol] = acc1[r] + cb_;
        }
    }
    __syncthreads();
    // LN stats: waves sweep rows (24 waves over 32 rows)
    for (int row = wave; row < 32; row += 24) {
        float v[12]; float s = 0.f;
        #pragma unroll
        for (int i = 0; i < 12; ++i) { v[i] = tile[row * C_DIM + lane + i * 32]; s += v[i]; }
        s = wave_red(s);
        const float mu = s / (float)C_DIM;
        float q = 0.f;
        #pragma unroll
        for (int i = 0; i < 12; ++i) { float d = v[i] - mu; q += d * d; }
        q = wave_red(q);
        if (lane == 0) { sMu[row] = mu; sRs[row] = rsqrtf(q / (float)C_DIM + LN_EPS); }
    }
    __syncthreads();
    // normalize + relu + packed f16 store: 16 elems per thread (32*384 total)
    {
        const int row = tid / 24;
        const int f   = (tid % 24) * 16;
        const float mu = sMu[row], rs = sRs[row];
        alignas(16) _Float16 o[16];
        #pragma unroll
        for (int j = 0; j < 16; ++j) {
            float val = (tile[row * C_DIM + f + j] - mu) * rs * lng[f + j] + lnb[f + j];
            o[j] = (_Float16)fmaxf(val, 0.f);
        }
        _Float16* op = out + ((size_t)b * Tp + 1 + t0 + row) * C_DIM + f;
        reinterpret_cast<uint4*>(op)[0] = reinterpret_cast<uint4*>(o)[0];
        reinterpret_cast<uint4*>(op)[1] = reinterpret_cast<uint4*>(o)[1];
    }
    // write this batch-row's pad rows (rows 0 and T+1), 48 uint4 each
    if (t0 == 0 && tid < 48)
        reinterpret_cast<uint4*>(out + (size_t)b * Tp * C_DIM)[tid] = make_uint4(0u, 0u, 0u, 0u);
    if (t0 + 32 == T && tid >= 48 && tid < 96)
        reinterpret_cast<uint4*>(out + ((size_t)b * Tp + T + 1) * C_DIM)[tid - 48] = make_uint4(0u, 0u, 0u, 0u);
}

// ---------------- fused conv + bias + LN + ReLU + dot(lw) + lb -> pred ----------------
// in: padded f16 [B][T+2][C]; pred: f32 [B][T]
__global__ __launch_bounds__(768)
void conv_ln_dot_kernel(const _Float16* __restrict__ in, const _Float16* __restrict__ wT,
                        const float* __restrict__ cbias, const float* __restrict__ lng,
                        const float* __restrict__ lnb, const float* __restrict__ lw,
                        const float* __restrict__ lb, float* __restrict__ pred, int T)
{
    __shared__ float tile[32 * C_DIM];
    __shared__ float sMu[32], sRs[32];
    const int tid   = threadIdx.x;
    const int wave  = tid >> 5;
    const int lane  = tid & 31;
    const int laneM = lane & 15;
    const int half  = lane >> 4;
    const int nb    = T / 32;
    const int b     = blockIdx.x / nb;
    const int t0    = (blockIdx.x % nb) * 32;
    const int f0    = wave * 16;
    const int Tp    = T + 2;

    v8f acc0 = {}, acc1 = {};
    #pragma unroll
    for (int tap = 0; tap < 3; ++tap) {
        const _Float16* pa0 = in + ((size_t)b * Tp + t0 + laneM + tap) * C_DIM + half * 8;
        const _Float16* pa1 = pa0 + (size_t)16 * C_DIM;
        const _Float16* pb  = wT + ((size_t)tap * C_DIM + f0 + laneM) * C_DIM + half * 8;
        for (int cb = 0; cb < C_DIM; cb += 32) {
            v16h a0, a1, bf;
            load_frag(a0, pa0 + cb);
            load_frag(a1, pa1 + cb);
            load_frag(bf, pb + cb);
            acc0 = __builtin_amdgcn_wmma_f32_16x16x32_f16(false, a0, false, bf,
                                                          (short)0, acc0, false, false);
            acc1 = __builtin_amdgcn_wmma_f32_16x16x32_f16(false, a1, false, bf,
                                                          (short)0, acc1, false, false);
        }
    }
    {
        const int fcol = f0 + laneM;
        const float cb_ = cbias[fcol];
        #pragma unroll
        for (int r = 0; r < 8; ++r) {
            const int m = r + half * 8;
            tile[m * C_DIM + fcol]        = acc0[r] + cb_;
            tile[(m + 16) * C_DIM + fcol] = acc1[r] + cb_;
        }
    }
    __syncthreads();
    for (int row = wave; row < 32; row += 24) {
        float v[12]; float s = 0.f;
        #pragma unroll
        for (int i = 0; i < 12; ++i) { v[i] = tile[row * C_DIM + lane + i * 32]; s += v[i]; }
        s = wave_red(s);
        const float mu = s / (float)C_DIM;
        float q = 0.f;
        #pragma unroll
        for (int i = 0; i < 12; ++i) { float d = v[i] - mu; q += d * d; }
        q = wave_red(q);
        if (lane == 0) { sMu[row] = mu; sRs[row] = rsqrtf(q / (float)C_DIM + LN_EPS); }
    }
    __syncthreads();
    // normalize + relu back into LDS (f32), 16 elems per thread
    {
        const int row = tid / 24;
        const int f   = (tid % 24) * 16;
        const float mu = sMu[row], rs = sRs[row];
        #pragma unroll
        for (int j = 0; j < 16; ++j) {
            float val = (tile[row * C_DIM + f + j] - mu) * rs * lng[f + j] + lnb[f + j];
            tile[row * C_DIM + f + j] = fmaxf(val, 0.f);
        }
    }
    __syncthreads();
    // per-row dot with lw
    for (int row = wave; row < 32; row += 24) {
        float s = 0.f;
        #pragma unroll
        for (int i = 0; i < 12; ++i)
            s += tile[row * C_DIM + lane + i * 32] * lw[lane + i * 32];
        s = wave_red(s);
        if (lane == 0) pred[(size_t)b * T + t0 + row] = s + lb[0];
    }
}

// ---------------- duration rounding ----------------
__global__ void dur_kernel(const float* __restrict__ logd, float* __restrict__ durr, int n) {
    int i = blockIdx.x * blockDim.x + threadIdx.x;
    if (i < n) durr[i] = rintf(fmaxf(expf(logd[i]) - 1.f, 0.f));
}

// ---------------- per-batch inclusive scan of duration_target ----------------
__global__ void scan_kernel(const int* __restrict__ dur, int* __restrict__ cum) {
    __shared__ int s[S_DIM];
    const int b = blockIdx.x, tid = threadIdx.x;
    s[tid] = dur[b * S_DIM + tid];
    __syncthreads();
    for (int off = 1; off < S_DIM; off <<= 1) {
        int v = (tid >= off) ? s[tid - off] : 0;
        __syncthreads();
        s[tid] += v;
        __syncthreads();
    }
    cum[b * S_DIM + tid] = s[tid];
}

// ---------------- length regulate: xe f32 [B][M][C] + xeb padded f16 [B][M+2][C] ----------------
__global__ void gather_kernel(const float* __restrict__ x, const int* __restrict__ cum,
                              float* __restrict__ xe, _Float16* __restrict__ xeb) {
    const int Mp = M_DIM + 2;
    const int b  = blockIdx.x / Mp;
    const int tp = blockIdx.x % Mp;
    if (tp == 0 || tp == M_DIM + 1) {           // pad rows: zero f16 only
        _Float16* rp = xeb + ((size_t)b * Mp + tp) * C_DIM;
        for (int f = threadIdx.x; f < C_DIM; f += blockDim.x) rp[f] = (_Float16)0.f;
        return;
    }
    const int t = tp - 1;
    const int* crow = cum + b * S_DIM;
    const int total = crow[S_DIM - 1];
    int lo = 0, hi = S_DIM;                      // searchsorted side='right'
    while (lo < hi) { int mid = (lo + hi) >> 1; if (crow[mid] <= t) lo = mid + 1; else hi = mid; }
    const int src = (lo < S_DIM - 1) ? lo : (S_DIM - 1);
    const float valid = (t < total) ? 1.f : 0.f;
    const float* xr = x + ((size_t)b * S_DIM + src) * H_DIM;
    const size_t oe = ((size_t)b * M_DIM + t) * H_DIM;
    const size_t ob = ((size_t)b * Mp + tp) * C_DIM;
    for (int f = threadIdx.x; f < H_DIM; f += blockDim.x) {
        const float v = xr[f] * valid;
        xe[oe + f]  = v;
        xeb[ob + f] = (_Float16)v;
    }
}

// ---------------- final combine: out = xe + mask*(p_emb + e_emb); mel_pos out ----------------
__global__ void combine_kernel(const float* __restrict__ xe, const int* __restrict__ mel_pos,
                               const float* __restrict__ ptgt, const float* __restrict__ etgt,
                               const float* __restrict__ pbins, const float* __restrict__ ebins,
                               const float* __restrict__ pemb, const float* __restrict__ eemb,
                               float* __restrict__ out, float* __restrict__ melOut) {
    const int b = blockIdx.x / M_DIM;
    const int t = blockIdx.x % M_DIM;
    const int mp = mel_pos[b * M_DIM + t];
    const float mask = (mp != 0) ? 1.f : 0.f;
    const float pv = ptgt[b * M_DIM + t];
    const float ev = etgt[b * M_DIM + t];
    int pi, ei;                                  // searchsorted side='left', 255 boundaries
    { int lo = 0, hi = NB_DIM - 1; while (lo < hi) { int m = (lo + hi) >> 1; if (pbins[m] < pv) lo = m + 1; else hi = m; } pi = lo; }
    { int lo = 0, hi = NB_DIM - 1; while (lo < hi) { int m = (lo + hi) >> 1; if (ebins[m] < ev) lo = m + 1; else hi = m; } ei = lo; }
    const size_t o = ((size_t)b * M_DIM + t) * H_DIM;
    for (int f = threadIdx.x; f < H_DIM; f += blockDim.x)
        out[o + f] = xe[o + f] + mask * (pemb[(size_t)pi * H_DIM + f] + eemb[(size_t)ei * H_DIM + f]);
    if (threadIdx.x == 0) melOut[b * M_DIM + t] = (float)mp;
}

// ---------------- host-side launch ----------------
extern "C" void kernel_launch(void* const* d_in, const int* in_sizes, int n_in,
                              void* d_out, int out_size, void* d_ws, size_t ws_size,
                              hipStream_t stream) {
    (void)in_sizes; (void)n_in; (void)out_size; (void)ws_size;
    // ---- inputs (setup_inputs dict order) ----
    const float* x     = (const float*)d_in[0];
    const int*   durT  = (const int*)d_in[1];
    const int*   melP  = (const int*)d_in[2];
    // d_in[3] = src_pos (unused)
    const float* ptgt  = (const float*)d_in[4];
    const float* etgt  = (const float*)d_in[5];
    const float* pbins = (const float*)d_in[36];
    const float* ebins = (const float*)d_in[37];
    const float* pemb  = (const float*)d_in[38];
    const float* eemb  = (const float*)d_in[39];

    // ---- d_out layout (return order, flat) ----
    float* out = (float*)d_out;
    const size_t O_LOGD = (size_t)B_DIM * M_DIM * H_DIM;
    const size_t O_DUR  = O_LOGD + (size_t)B_DIM * S_DIM;
    const size_t O_PP   = O_DUR  + (size_t)B_DIM * S_DIM;
    const size_t O_EP   = O_PP   + (size_t)B_DIM * M_DIM;
    const size_t O_MEL  = O_EP   + (size_t)B_DIM * M_DIM;

    // ---- workspace carve-up (256B aligned sizes) ----
    char* ws = (char*)d_ws;
    size_t off = 0;
    auto take = [&](size_t bytes) { char* p = ws + off; off += (bytes + 255) & ~(size_t)255; return p; };
    _Float16* xb   = (_Float16*)take((size_t)B_DIM * (S_DIM + 2) * C_DIM * 2);   // padded x f16
    _Float16* wTd1 = (_Float16*)take((size_t)3 * C_DIM * C_DIM * 2);
    _Float16* wTd2 = (_Float16*)take((size_t)3 * C_DIM * C_DIM * 2);
    _Float16* wTp1 = (_Float16*)take((size_t)3 * C_DIM * C_DIM * 2);
    _Float16* wTp2 = (_Float16*)take((size_t)3 * C_DIM * C_DIM * 2);
    _Float16* wTe1 = (_Float16*)take((size_t)3 * C_DIM * C_DIM * 2);
    _Float16* wTe2 = (_Float16*)take((size_t)3 * C_DIM * C_DIM * 2);
    int*      cum  = (int*)take((size_t)B_DIM * S_DIM * 4);
    float*    xe   = (float*)take((size_t)B_DIM * M_DIM * H_DIM * 4);
    _Float16* xeb  = (_Float16*)take((size_t)B_DIM * (M_DIM + 2) * C_DIM * 2);   // padded
    _Float16* h1   = (_Float16*)take((size_t)B_DIM * (M_DIM + 2) * C_DIM * 2);   // padded, reused

    const int nW  = 3 * C_DIM * C_DIM;
    const int nXp = B_DIM * (S_DIM + 2) * C_DIM;

    // 1) precision prep
    cvt_pad_kernel<<<(nXp + 255) / 256, 256, 0, stream>>>(x, xb, S_DIM);
    transpose_w_kernel<<<(nW + 255) / 256, 256, 0, stream>>>((const float*)d_in[6],  wTd1);
    transpose_w_kernel<<<(nW + 255) / 256, 256, 0, stream>>>((const float*)d_in[10], wTd2);
    transpose_w_kernel<<<(nW + 255) / 256, 256, 0, stream>>>((const float*)d_in[16], wTp1);
    transpose_w_kernel<<<(nW + 255) / 256, 256, 0, stream>>>((const float*)d_in[20], wTp2);
    transpose_w_kernel<<<(nW + 255) / 256, 256, 0, stream>>>((const float*)d_in[26], wTe1);
    transpose_w_kernel<<<(nW + 255) / 256, 256, 0, stream>>>((const float*)d_in[30], wTe2);

    // 2) duration predictor on x  (T = S = 256)
    conv_ln_relu_kernel<<<B_DIM * (S_DIM / 32), 768, 0, stream>>>(
        xb, wTd1, (const float*)d_in[7], (const float*)d_in[8], (const float*)d_in[9], h1, S_DIM);
    conv_ln_dot_kernel<<<B_DIM * (S_DIM / 32), 768, 0, stream>>>(
        h1, wTd2, (const float*)d_in[11], (const float*)d_in[12], (const float*)d_in[13],
        (const float*)d_in[14], (const float*)d_in[15], out + O_LOGD, S_DIM);
    dur_kernel<<<(B_DIM * S_DIM + 255) / 256, 256, 0, stream>>>(out + O_LOGD, out + O_DUR, B_DIM * S_DIM);

    // 3) length regulate
    scan_kernel<<<B_DIM, S_DIM, 0, stream>>>(durT, cum);
    gather_kernel<<<B_DIM * (M_DIM + 2), 128, 0, stream>>>(x, cum, xe, xeb);

    // 4) pitch predictor on xe  (T = M = 2048)
    conv_ln_relu_kernel<<<B_DIM * (M_DIM / 32), 768, 0, stream>>>(
        xeb, wTp1, (const float*)d_in[17], (const float*)d_in[18], (const float*)d_in[19], h1, M_DIM);
    conv_ln_dot_kernel<<<B_DIM * (M_DIM / 32), 768, 0, stream>>>(
        h1, wTp2, (const float*)d_in[21], (const float*)d_in[22], (const float*)d_in[23],
        (const float*)d_in[24], (const float*)d_in[25], out + O_PP, M_DIM);

    // 5) energy predictor on xe
    conv_ln_relu_kernel<<<B_DIM * (M_DIM / 32), 768, 0, stream>>>(
        xeb, wTe1, (const float*)d_in[27], (const float*)d_in[28], (const float*)d_in[29], h1, M_DIM);
    conv_ln_dot_kernel<<<B_DIM * (M_DIM / 32), 768, 0, stream>>>(
        h1, wTe2, (const float*)d_in[31], (const float*)d_in[32], (const float*)d_in[33],
        (const float*)d_in[34], (const float*)d_in[35], out + O_EP, M_DIM);

    // 6) combine
    combine_kernel<<<B_DIM * M_DIM, 128, 0, stream>>>(
        xe, melP, ptgt, etgt, pbins, ebins, pemb, eemb, out, out + O_MEL);
}
